// SMP_LineGraph_26998164423029
// MI455X (gfx1250) — matrix operand order repaired and checked by reading, not compiled, not run on hardware
//
#include <hip/hip_runtime.h>

// ---------------- problem constants (uniform-graph structure) ----------------
typedef __bf16 bf16;
typedef __attribute__((ext_vector_type(16))) bf16  v16bf;
typedef __attribute__((ext_vector_type(8)))  float v8f;

constexpr int G_    = 128;
constexpr int NV_   = 32;
constexpr int EPG_  = 64;
constexpr int LEPG_ = 256;
constexpr int NG_   = G_ * NV_;    // 4096
constexpr int NLG_  = G_ * EPG_;   // 8192
constexpr int ELG_  = G_ * LEPG_;  // 32768
constexpr int K_    = 64;          // colors per graph
constexpr int H_    = 32;
constexpr int HF_   = 128;
constexpr int L_    = 4;
constexpr int NC_   = 12;
constexpr int FIN_  = 16;
constexpr int FE_   = 16;
constexpr int NB_   = 4;
constexpr long UR_  = (long)NLG_ * K_;  // 524288 rows of u

// ---------------- WMMA helpers (CDNA5 16x16x32 bf16 -> f32) ----------------
__device__ __forceinline__ v8f wmma_bf16(v16bf a, v16bf b, v8f c) {
  return __builtin_amdgcn_wmma_f32_16x16x32_bf16(false, a, false, b, (short)0, c,
                                                 false, false);
}

// A fragment: 16x32 tile, row-major f32 source (ISA 16-bit A layout:
// lanes 0-15 hold K={0..7,16..23}, lanes 16-31 hold K={8..15,24..31}).
__device__ __forceinline__ v16bf frag_a_f32(const float* base, int ld, int lane) {
  int r = lane & 15, hi = (lane >> 4) * 8;
  const float* row = base + r * ld;
  v16bf a;
#pragma unroll
  for (int i = 0; i < 8; ++i) { a[i] = (bf16)row[hi + i]; a[i + 8] = (bf16)row[16 + hi + i]; }
  return a;
}
__device__ __forceinline__ v16bf frag_a_bf16lds(const bf16* base, int ld, int lane) {
  int r = lane & 15, hi = (lane >> 4) * 8;
  const bf16* row = base + r * ld;
  v16bf a;
#pragma unroll
  for (int i = 0; i < 8; ++i) { a[i] = row[hi + i]; a[i + 8] = row[16 + hi + i]; }
  return a;
}
// B fragment: 32x16 tile from K-major f32 W[k*ld + n]; zero-pad k >= kmax.
// (lanes 0-15: K={0..7,16..23}, lanes 16-31: K={8..15,24..31}, N = lane&15)
__device__ __forceinline__ v16bf frag_b_f32(const float* base, int ld, int kmax, int lane) {
  int n = lane & 15, k0 = (lane >> 4) * 8;
  v16bf b;
#pragma unroll
  for (int i = 0; i < 16; ++i) {
    int k = k0 + i + (i & 8);
    b[i] = (k < kmax) ? (bf16)base[k * ld + n] : (bf16)0.0f;
  }
  return b;
}
// B fragment from dense 32x32 bf16 weight in LDS, column tile n0 in {0,16}.
__device__ __forceinline__ v16bf frag_b_lds(const bf16* Wd, int n0, int lane) {
  int n = lane & 15, k0 = (lane >> 4) * 8;
  v16bf b;
#pragma unroll
  for (int i = 0; i < 16; ++i) b[i] = Wd[(k0 + i + (i & 8)) * 32 + n0 + n];
  return b;
}

// ---------------- small generic kernels ----------------
__global__ void k_zero_i(int* p, int n) {
  for (int i = blockIdx.x * blockDim.x + threadIdx.x; i < n; i += gridDim.x * blockDim.x) p[i] = 0;
}
__global__ void k_zero_f(float* p, int n) {
  for (int i = blockIdx.x * blockDim.x + threadIdx.x; i < n; i += gridDim.x * blockDim.x) p[i] = 0.f;
}
// Y[M,N] = X[M,Kin] @ W[Kin,N] + b
__global__ void k_lin(const float* X, const float* W, const float* b, float* Y,
                      int M, int Kin, int N) {
  int idx = blockIdx.x * blockDim.x + threadIdx.x;
  if (idx >= M * N) return;
  int m = idx / N, n = idx - m * N;
  float acc = b[n];
  for (int k = 0; k < Kin; ++k) acc += X[m * Kin + k] * W[k * N + n];
  Y[idx] = acc;
}
__global__ void k_color(const int* eig, const int* batch, int* color) {
  int n = blockIdx.x * blockDim.x + threadIdx.x;
  if (n < NLG_) color[n] = n - batch[eig[n]] * EPG_;
}

// ---------------- CSR build for lg-edge destinations ----------------
__global__ void k_count(const int* eil, int* cnt) {
  int e = blockIdx.x * blockDim.x + threadIdx.x;
  if (e < ELG_) atomicAdd(&cnt[eil[ELG_ + e]], 1);
}
__global__ void k_scan(const int* cnt, int* offs, int* cursor) {
  __shared__ int ps[1024];
  int t = threadIdx.x;                 // 1024 threads, 8 counts each
  int base = t * 8, loc[8], s = 0;
#pragma unroll
  for (int j = 0; j < 8; ++j) { loc[j] = s; s += cnt[base + j]; }
  ps[t] = s;
  __syncthreads();
  for (int off = 1; off < 1024; off <<= 1) {
    int v = (t >= off) ? ps[t - off] : 0;
    __syncthreads();
    ps[t] += v;
    __syncthreads();
  }
  int excl = (t == 0) ? 0 : ps[t - 1];
#pragma unroll
  for (int j = 0; j < 8; ++j) { int o = excl + loc[j]; offs[base + j] = o; cursor[base + j] = o; }
  if (t == 1023) offs[NLG_] = ps[1023];
}
__global__ void k_fill(const int* eil, int* cursor, int* csre) {
  int e = blockIdx.x * blockDim.x + threadIdx.x;
  if (e < ELG_) { int pos = atomicAdd(&cursor[eil[ELG_ + e]], 1); csre[pos] = e; }
}

// ---------------- hm = concat(h[s],h[d],ea_g,x_lg) @ W_msg + b  (WMMA) ------
// block: 128 thr = 4 waves; block -> 16 rows, wave w -> cols [32w,32w+32); K=84 pad 96
__global__ void k_msg(const float* h, const float* eag, const float* xlg,
                      const int* eig, const float* Wmsg, const float* bmsg, float* hm) {
  __shared__ bf16 A[16][96];
  int row0 = blockIdx.x * 16, tid = threadIdx.x;
  for (int idx = tid; idx < 16 * 96; idx += 128) {
    int r = idx / 96, c = idx - r * 96, n = row0 + r;
    float v;
    if (c < 32)      v = h[eig[n] * H_ + c];
    else if (c < 64) v = h[eig[NLG_ + n] * H_ + (c - 32)];
    else if (c < 80) v = eag[n * FE_ + (c - 64)];
    else if (c < 84) v = xlg[n * NB_ + (c - 80)];
    else             v = 0.f;
    A[r][c] = (bf16)v;
  }
  __syncthreads();
  int wave = tid >> 5, lane = tid & 31, n0 = wave * 32;
  v8f acc0 = {}, acc1 = {};
#pragma unroll
  for (int kc = 0; kc < 3; ++kc) {
    v16bf a  = frag_a_bf16lds(&A[0][kc * 32], 96, lane);
    v16bf b0 = frag_b_f32(Wmsg + kc * 32 * HF_ + n0,      HF_, 84 - kc * 32, lane);
    v16bf b1 = frag_b_f32(Wmsg + kc * 32 * HF_ + n0 + 16, HF_, 84 - kc * 32, lane);
    acc0 = wmma_bf16(a, b0, acc0);
    acc1 = wmma_bf16(a, b1, acc1);
  }
  int nn = lane & 15, mb = (lane >> 4) * 8;
#pragma unroll
  for (int j = 0; j < 8; ++j) {
    int r = row0 + mb + j;
    hm[r * HF_ + n0 + nn]      = acc0[j] + bmsg[n0 + nn];
    hm[r * HF_ + n0 + 16 + nn] = acc1[j] + bmsg[n0 + 16 + nn];
  }
}

// ---------------- first extract (uses sparse structure of u0) --------------
__global__ void k_graphsum(const float* hm, float* S) {   // S[g][c] = sum_n hm
  int g = blockIdx.x, c = threadIdx.x;   // 128 threads
  float s = 0.f;
  for (int j = 0; j < EPG_; ++j) s += hm[(g * EPG_ + j) * HF_ + c];
  S[g * HF_ + c] = s;
}
__global__ void k_out0(const float* S, const float* w1, const float* b1,
                       const float* w2, const float* b2, const float* lw,
                       const float* lb, float* out) {
  __shared__ float oe[HF_];
  int g = blockIdx.x, t = threadIdx.x;   // 128 threads
  const float i4096 = 1.f / 4096.f, i64 = 1.f / 64.f;
  float v = 64.f * (w1[t] * i4096 + w2[t] * i64) + b1[t] + b2[t];  // one-hot channel (=64)
  for (int c = 0; c < HF_; ++c) {
    float s = S[g * HF_ + c];
    v += s * (w1[(c + 1) * HF_ + t] * i4096 + w2[(c + 1) * HF_ + t] * i64);
  }
  oe[t] = v;
  __syncthreads();
  float r = lb[t];
  for (int c = 0; c < HF_; ++c) r += oe[c] * lw[c * HF_ + t];
  out[g * HF_ + t] = oe[t] + fmaxf(r, 0.f);
}

// ---------------- t[n] = W_init[0] + hm[n] @ W_init[1:]  (WMMA) ------------
// block: 128 thr = 4 waves; block -> 64 rows (wave -> 16 rows, all 32 cols), K=128
__global__ void k_uinit(const float* hm, const float* Winit, float* tvec) {
  __shared__ bf16 A[64][128];
  int row0 = blockIdx.x * 64, tid = threadIdx.x;
  for (int idx = tid; idx < 64 * 128; idx += 128) {
    int r = idx >> 7, c = idx & 127;
    A[r][c] = (bf16)hm[(row0 + r) * HF_ + c];
  }
  __syncthreads();
  int wave = tid >> 5, lane = tid & 31;
  v8f acc0 = {}, acc1 = {};
#pragma unroll
  for (int kc = 0; kc < 4; ++kc) {
    v16bf a  = frag_a_bf16lds(&A[wave * 16][kc * 32], 128, lane);
    v16bf b0 = frag_b_f32(Winit + H_ + kc * 32 * H_,      H_, 32, lane);
    v16bf b1 = frag_b_f32(Winit + H_ + kc * 32 * H_ + 16, H_, 32, lane);
    acc0 = wmma_bf16(a, b0, acc0);
    acc1 = wmma_bf16(a, b1, acc1);
  }
  int nn = lane & 15, mb = (lane >> 4) * 8;
#pragma unroll
  for (int j = 0; j < 8; ++j) {
    int r = row0 + wave * 16 + mb + j;
    tvec[r * H_ + nn]      = acc0[j] + Winit[nn];
    tvec[r * H_ + 16 + nn] = acc1[j] + Winit[16 + nn];
  }
}
__global__ void k_uscatter(const float* tvec, const float* binit, const int* color, float* u) {
  for (long idx = blockIdx.x * (long)blockDim.x + threadIdx.x; idx < UR_ * H_;
       idx += (long)gridDim.x * blockDim.x) {
    int c = idx & 31;
    long row = idx >> 5;
    int k = row & 63, n = (int)(row >> 6);
    float v = binit[c];
    if (k == color[n]) v += tvec[n * H_ + c];
    u[idx] = v;
  }
}

// ---------------- batch norm (deterministic two-stage tree) ----------------
__global__ void k_bnstats(const float* u, float* part) {   // 128 blocks x 256
  int t = threadIdx.x, c = t & 31, rl = t >> 5;            // 8 row-lanes x 32 ch
  float s = 0.f, s2 = 0.f;
  for (long row = blockIdx.x * 8 + rl; row < UR_; row += (long)gridDim.x * 8) {
    float v = u[row * H_ + c];
    s += v; s2 += v * v;
  }
  __shared__ float sh1[8][32], sh2[8][32];
  sh1[rl][c] = s; sh2[rl][c] = s2;
  __syncthreads();
  if (t < 32) {
    float a = 0.f, b = 0.f;
    for (int j = 0; j < 8; ++j) { a += sh1[j][t]; b += sh2[j][t]; }
    part[blockIdx.x * 64 + t] = a;
    part[blockIdx.x * 64 + 32 + t] = b;
  }
}
__global__ void k_bnfinal(const float* part, const float* bng, const float* bnb,
                          float* scsh, int apply) {
  int c = threadIdx.x;  // 32
  if (c >= 32) return;
  float sc = 1.f, sh = 0.f;
  if (apply) {
    float s = 0.f, s2 = 0.f;
    for (int b = 0; b < 128; ++b) { s += part[b * 64 + c]; s2 += part[b * 64 + 32 + c]; }
    const float inv = 1.f / (float)UR_;
    float m = s * inv, var = s2 * inv - m * m;
    sc = bng[c] * rsqrtf(var + 1e-5f);
    sh = bnb[c] - m * sc;
  }
  scsh[c] = sc; scsh[32 + c] = sh;
}

// ---------------- per-node: usum/udiag -> pvec; dist_loc -------------------
__global__ void k_nodepre(const float* u, const float* scsh, const int* color,
                          const float* u1b, const float* u2w, const float* u2b,
                          const float* u3w, const float* u3b, const float* dglob,
                          const float* ldw, const float* ldb, float* pvec, float* dloc) {
  int n = blockIdx.x, c = threadIdx.x;  // 32 threads (one wave)
  float sc = scsh[c], sh = scsh[32 + c];
  const float* un = u + (long)n * K_ * H_;
  float s = 0.f;
  for (int k = 0; k < K_; ++k) s += un[k * H_ + c];
  s = (s * sc + 64.f * sh) * (1.f / 64.f);
  float dg = un[color[n] * H_ + c] * sc + sh;
  __shared__ float su[32], di[32];
  su[c] = s; di[c] = dg;
  __syncthreads();
  int tw = c >> 2, co = c & 3;
  float pv = u1b[c] + u2b[c] + u3b[c];
#pragma unroll
  for (int ci = 0; ci < 4; ++ci) {
    pv += su[tw * 4 + ci] * u2w[tw * 16 + ci * 4 + co];
    pv += di[tw * 4 + ci] * u3w[tw * 16 + ci * 4 + co];
  }
  pvec[n * H_ + c] = pv;
  float dl = ldb[c];
#pragma unroll
  for (int b = 0; b < 4; ++b) dl += dglob[n * 4 + b] * ldw[b * H_ + c];
  dloc[n * H_ + c] = dl;
}

// ---------------- um = (u_bn@W1 + pvec)*dist ; ui = um@Wli ; uj = um@Wlj ---
// 256 thr = 8 waves; wave -> one 16-row tile (tile lies inside one node). WMMA x6.
__global__ void k_um(float* u, const float* scsh, const float* pvec, const float* dloc,
                     const float* w1, const float* wli, const float* wlj,
                     const float* lib, const float* ljb,
                     bf16* umg, bf16* uig, bf16* ujg) {
  __shared__ bf16 W1d[1024], WId[1024], WJd[1024];
  __shared__ float stage[8][16][32];
  int tid = threadIdx.x;
  for (int idx = tid; idx < 1024; idx += 256) {     // expand block-diag towers
    int r = idx >> 5, c = idx & 31;
    bool on = (r >> 2) == (c >> 2);
    int wi = (r >> 2) * 16 + (r & 3) * 4 + (c & 3);
    W1d[idx] = (bf16)(on ? w1[wi] : 0.f);
    WId[idx] = (bf16)(on ? wli[wi] : 0.f);
    WJd[idx] = (bf16)(on ? wlj[wi] : 0.f);
  }
  __syncthreads();
  int wave = tid >> 5, lane = tid & 31;
  long row0 = (long)blockIdx.x * 128 + wave * 16;
  int node = (int)(row0 >> 6);
  int r = lane & 15, hi = (lane >> 4) * 8;
  // load u rows, apply bn, write bn'd u back in place
  float* urow = u + (row0 + r) * H_;
  float av[16];
#pragma unroll
  for (int i = 0; i < 8; ++i) {
    av[i]     = urow[hi + i]      * scsh[hi + i]      + scsh[32 + hi + i];
    av[i + 8] = urow[16 + hi + i] * scsh[16 + hi + i] + scsh[48 + hi + i];
  }
#pragma unroll
  for (int i = 0; i < 8; ++i) { urow[hi + i] = av[i]; urow[16 + hi + i] = av[i + 8]; }
  v16bf a;
#pragma unroll
  for (int i = 0; i < 16; ++i) a[i] = (bf16)av[i];
  int nn = lane & 15, mb = (lane >> 4) * 8;
  float pv0 = pvec[node * H_ + nn], pv1 = pvec[node * H_ + 16 + nn];
  float dl0 = dloc[node * H_ + nn], dl1 = dloc[node * H_ + 16 + nn];
  v8f acc0, acc1;
#pragma unroll
  for (int j = 0; j < 8; ++j) { acc0[j] = pv0; acc1[j] = pv1; }
  acc0 = wmma_bf16(a, frag_b_lds(W1d, 0, lane), acc0);
  acc1 = wmma_bf16(a, frag_b_lds(W1d, 16, lane), acc1);
#pragma unroll
  for (int j = 0; j < 8; ++j) { acc0[j] *= dl0; acc1[j] *= dl1; }
#pragma unroll
  for (int j = 0; j < 8; ++j) {                         // store um (bf16)
    long rr = row0 + mb + j;
    umg[rr * H_ + nn]      = (bf16)acc0[j];
    umg[rr * H_ + 16 + nn] = (bf16)acc1[j];
    stage[wave][mb + j][nn]      = acc0[j];             // relayout C->A via LDS
    stage[wave][mb + j][16 + nn] = acc1[j];
  }
  __syncthreads();
  const float* sp = &stage[wave][0][0];
  v16bf am;
#pragma unroll
  for (int i = 0; i < 8; ++i) { am[i] = (bf16)sp[r * 32 + hi + i]; am[i + 8] = (bf16)sp[r * 32 + 16 + hi + i]; }
  float bi0 = lib[nn], bi1 = lib[16 + nn], bj0 = ljb[nn], bj1 = ljb[16 + nn];
  v8f ai0, ai1, aj0, aj1;
#pragma unroll
  for (int j = 0; j < 8; ++j) { ai0[j] = bi0; ai1[j] = bi1; aj0[j] = bj0; aj1[j] = bj1; }
  ai0 = wmma_bf16(am, frag_b_lds(WId, 0, lane), ai0);
  ai1 = wmma_bf16(am, frag_b_lds(WId, 16, lane), ai1);
  aj0 = wmma_bf16(am, frag_b_lds(WJd, 0, lane), aj0);
  aj1 = wmma_bf16(am, frag_b_lds(WJd, 16, lane), aj1);
#pragma unroll
  for (int j = 0; j < 8; ++j) {
    long rr = row0 + mb + j;
    uig[rr * H_ + nn]      = (bf16)ai0[j];
    uig[rr * H_ + 16 + nn] = (bf16)ai1[j];
    ujg[rr * H_ + nn]      = (bf16)aj0[j];
    ujg[rr * H_ + 16 + nn] = (bf16)aj1[j];
  }
}

// ---------------- ef = (eglob@la + lab)@en + enb ---------------------------
__global__ void k_ef(const float* eglob, const float* law, const float* lab,
                     const float* enw, const float* enb, float* efv) {
  int idx = blockIdx.x * blockDim.x + threadIdx.x;
  if (idx >= ELG_ * H_) return;
  int e = idx >> 5, c = idx & 31;
  float acc = enb[c];
  for (int j = 0; j < H_; ++j) {
    float el = lab[j];
#pragma unroll
    for (int b = 0; b < 4; ++b) el += eglob[e * 4 + b] * law[b * H_ + j];
    acc += el * enw[j * H_ + c];
  }
  efv[idx] = acc;
}

// ---------------- CSR aggregate: u[n] += 1/4 * sum_in msg ------------------
__global__ void k_msgagg(float* u, const bf16* umg, const bf16* uig, const bf16* ujg,
                         const float* efv, const int* offs, const int* csre, const int* eil) {
  int n = blockIdx.x;
  int beg = offs[n], end = offs[n + 1];
  float* un = u + (long)n * 2048;
  const bf16* uin = uig + (long)n * 2048;
  for (int idx = threadIdx.x; idx < 2048; idx += 256) {
    int c = idx & 31;
    float uival = (float)uin[idx];
    float acc = un[idx];
    for (int t = beg; t < end; ++t) {
      int e = csre[t];
      long src = eil[e];   // ls
      float m = (float)umg[src * 2048 + idx] +
                fmaxf(uival + (float)ujg[src * 2048 + idx] + efv[e * H_ + c], 0.f);
      acc += 0.25f * m;
    }
    un[idx] = acc;
  }
}

// ---------------- per-layer extract into out -------------------------------
__global__ void k_extract(const float* u, const int* color, const float* w1,
                          const float* b1, const float* w2, const float* b2,
                          const float* lw, const float* lb, float* out) {
  int g = blockIdx.x, tid = threadIdx.x, c = tid & 31, sl = tid >> 5;
  float macc = 0.f, dacc = 0.f;
  for (int lr = sl; lr < 4096; lr += 8) macc += u[((long)g * 4096 + lr) * H_ + c];
  for (int ln = sl; ln < 64; ln += 8) {
    int n = g * 64 + ln;
    dacc += u[((long)n * 64 + color[n]) * H_ + c];
  }
  __shared__ float ms[8][32], dsh[8][32], mv[32], dv[32], oe[HF_];
  ms[sl][c] = macc; dsh[sl][c] = dacc;
  __syncthreads();
  if (tid < 32) {
    float a = 0.f, b = 0.f;
    for (int j = 0; j < 8; ++j) { a += ms[j][tid]; b += dsh[j][tid]; }
    mv[tid] = a * (1.f / 4096.f); dv[tid] = b * (1.f / 64.f);
  }
  __syncthreads();
  if (tid < HF_) {
    float v = b1[tid] + b2[tid];
    for (int k = 0; k < H_; ++k) v += mv[k] * w1[k * HF_ + tid] + dv[k] * w2[k * HF_ + tid];
    oe[tid] = v;
  }
  __syncthreads();
  if (tid < HF_) {
    float r = lb[tid];
    for (int k = 0; k < HF_; ++k) r += oe[k] * lw[k * HF_ + tid];
    out[g * HF_ + tid] += 0.25f * (oe[tid] + fmaxf(r, 0.f));
  }
}

// ---------------- head: out = relu(out@ac)+out ; d_out = out@fl ------------
__global__ void k_final(const float* out, const float* acw, const float* acb,
                        const float* flw, const float* flb, float* dout) {
  __shared__ float o2[HF_];
  int g = blockIdx.x, t = threadIdx.x;  // 128
  float r = acb[t];
  for (int c = 0; c < HF_; ++c) r += out[g * HF_ + c] * acw[c * HF_ + t];
  o2[t] = out[g * HF_ + t] + fmaxf(r, 0.f);
  __syncthreads();
  if (t < NC_) {
    float v = flb[t];
    for (int c = 0; c < HF_; ++c) v += o2[c] * flw[c * NC_ + t];
    dout[g * NC_ + t] = v;
  }
}

// ---------------- launch ----------------------------------------------------
extern "C" void kernel_launch(void* const* d_in, const int* in_sizes, int n_in,
                              void* d_out, int out_size, void* d_ws, size_t ws_size,
                              hipStream_t stream) {
  (void)in_sizes; (void)n_in; (void)out_size; (void)ws_size;
  const float* x_g  = (const float*)d_in[0];
  const float* eag  = (const float*)d_in[1];
  const float* xlg  = (const float*)d_in[2];
  const float* ealg = (const float*)d_in[3];
  const float* Wenc = (const float*)d_in[4];
  const float* benc = (const float*)d_in[5];
  const float* Wmsg = (const float*)d_in[6];
  const float* bmsg = (const float*)d_in[7];
  const float* Winit= (const float*)d_in[8];
  const float* binit= (const float*)d_in[9];
  const float* np1w = (const float*)d_in[10]; const float* np1b = (const float*)d_in[11];
  const float* np2w = (const float*)d_in[12]; const float* np2b = (const float*)d_in[13];
  const float* nplw = (const float*)d_in[14]; const float* nplb = (const float*)d_in[15];
  const float* fe1w = (const float*)d_in[16]; const float* fe1b = (const float*)d_in[17];
  const float* fe2w = (const float*)d_in[18]; const float* fe2b = (const float*)d_in[19];
  const float* felw = (const float*)d_in[20]; const float* felb = (const float*)d_in[21];
  const float* gdw  = (const float*)d_in[22]; const float* gdb  = (const float*)d_in[23];
  const float* gaw  = (const float*)d_in[24]; const float* gab  = (const float*)d_in[25];
  const float* ldw  = (const float*)d_in[26]; const float* ldbv = (const float*)d_in[27];
  const float* law  = (const float*)d_in[28]; const float* labv = (const float*)d_in[29];
  const float* u1w  = (const float*)d_in[30]; const float* u1b  = (const float*)d_in[31];
  const float* u2w  = (const float*)d_in[32]; const float* u2b  = (const float*)d_in[33];
  const float* u3w  = (const float*)d_in[34]; const float* u3b  = (const float*)d_in[35];
  const float* enw  = (const float*)d_in[36]; const float* enb  = (const float*)d_in[37];
  const float* liw  = (const float*)d_in[38]; const float* lib  = (const float*)d_in[39];
  const float* ljw  = (const float*)d_in[40]; const float* ljb  = (const float*)d_in[41];
  const float* bng  = (const float*)d_in[42]; const float* bnb  = (const float*)d_in[43];
  const float* acw  = (const float*)d_in[44]; const float* acb  = (const float*)d_in[45];
  const float* flw  = (const float*)d_in[46]; const float* flb  = (const float*)d_in[47];
  const int* eig    = (const int*)d_in[48];
  const int* eil    = (const int*)d_in[49];
  const int* batch  = (const int*)d_in[50];

  char* p = (char*)d_ws;
  auto alloc = [&](size_t bytes) { char* r = p; p += (bytes + 255) & ~(size_t)255; return r; };
  float* h     = (float*)alloc((size_t)NG_ * H_ * 4);
  float* hm    = (float*)alloc((size_t)NLG_ * HF_ * 4);
  float* u     = (float*)alloc((size_t)UR_ * H_ * 4);
  bf16*  umg   = (bf16*)alloc((size_t)UR_ * H_ * 2);
  bf16*  uig   = (bf16*)alloc((size_t)UR_ * H_ * 2);
  bf16*  ujg   = (bf16*)alloc((size_t)UR_ * H_ * 2);
  float* tvec  = (float*)alloc((size_t)NLG_ * H_ * 4);
  float* pvec  = (float*)alloc((size_t)NLG_ * H_ * 4);
  float* dloc  = (float*)alloc((size_t)NLG_ * H_ * 4);
  float* efv   = (float*)alloc((size_t)ELG_ * H_ * 4);
  float* dglob = (float*)alloc((size_t)NLG_ * NB_ * 4);
  float* eglob = (float*)alloc((size_t)ELG_ * NB_ * 4);
  float* Sg    = (float*)alloc((size_t)G_ * HF_ * 4);
  float* outg  = (float*)alloc((size_t)G_ * HF_ * 4);
  float* bnpart= (float*)alloc(128 * 64 * 4);
  float* scsh  = (float*)alloc(64 * 4);
  int*   color = (int*)alloc(NLG_ * 4);
  int*   cnt   = (int*)alloc(NLG_ * 4);
  int*   offs  = (int*)alloc((NLG_ + 1) * 4);
  int*   cursor= (int*)alloc(NLG_ * 4);
  int*   csre  = (int*)alloc(ELG_ * 4);
  float* dout  = (float*)d_out;

  // small linears + structure
  k_lin<<<(NG_ * H_ + 255) / 256, 256, 0, stream>>>(x_g, Wenc, benc, h, NG_, FIN_, H_);
  k_lin<<<(NLG_ * NB_ + 255) / 256, 256, 0, stream>>>(xlg, gdw, gdb, dglob, NLG_, NB_, NB_);
  k_lin<<<(ELG_ * NB_ + 255) / 256, 256, 0, stream>>>(ealg, gaw, gab, eglob, ELG_, NB_, NB_);
  k_color<<<NLG_ / 256, 256, 0, stream>>>(eig, batch, color);
  k_zero_i<<<32, 256, 0, stream>>>(cnt, NLG_);
  k_count<<<ELG_ / 256, 256, 0, stream>>>(eil, cnt);
  k_scan<<<1, 1024, 0, stream>>>(cnt, offs, cursor);
  k_fill<<<ELG_ / 256, 256, 0, stream>>>(eil, cursor, csre);

  // hm (WMMA), first extract, u init (WMMA)
  k_msg<<<NLG_ / 16, 128, 0, stream>>>(h, eag, xlg, eig, Wmsg, bmsg, hm);
  k_graphsum<<<G_, 128, 0, stream>>>(hm, Sg);
  k_out0<<<G_, 128, 0, stream>>>(Sg, np1w, np1b, np2w, np2b, nplw, nplb, outg);
  k_uinit<<<NLG_ / 64, 128, 0, stream>>>(hm, Winit, tvec);
  k_uscatter<<<8192, 256, 0, stream>>>(tvec, binit, color, u);

  for (int i = 0; i < L_; ++i) {
    if (i > 0) k_bnstats<<<128, 256, 0, stream>>>(u, bnpart);
    k_bnfinal<<<1, 32, 0, stream>>>(bnpart, bng + i * H_, bnb + i * H_, scsh, i > 0 ? 1 : 0);
    k_nodepre<<<NLG_, 32, 0, stream>>>(u, scsh, color, u1b + i * H_, u2w + i * 128,
                                       u2b + i * H_, u3w + i * 128, u3b + i * H_,
                                       dglob, ldw + i * 128, ldbv + i * H_, pvec, dloc);
    k_um<<<(int)(UR_ / 128), 256, 0, stream>>>(u, scsh, pvec, dloc, u1w + i * 128,
                                               liw + i * 128, ljw + i * 128,
                                               lib + i * H_, ljb + i * H_, umg, uig, ujg);
    k_ef<<<(ELG_ * H_) / 256, 256, 0, stream>>>(eglob, law + i * 128, labv + i * H_,
                                                enw + i * 1024, enb + i * H_, efv);
    k_msgagg<<<NLG_, 256, 0, stream>>>(u, umg, uig, ujg, efv, offs, csre, eil);
    k_extract<<<G_, 256, 0, stream>>>(u, color, fe1w, fe1b, fe2w, fe2b, felw, felb, outg);
  }
  k_final<<<G_, 128, 0, stream>>>(outg, acw, acb, flw, flb, dout);
}
